// FlexibleScanpathHistoryEncoding_22986664968791
// MI455X (gfx1250) — matrix operand order repaired and checked by reading, not compiled, not run on hardware
//
#include <hip/hip_runtime.h>

// CDNA5 / gfx1250, wave32. Grouped 3x3 VALID conv as tiled GEMM on
// V_WMMA_F32_16X16X32_BF16 with a bf16x3 split (x=xh+xl, w=wh+wl,
// acc += wh*xh + wh*xl + wl*xh in f32) for near-fp32 accuracy at the
// 16-bit matrix-pipe rate (16384 FLOP/instr vs 2048 for f32 WMMA).
//
// GEMM view per (batch, output row, 64-pixel tile):
//   D[oc, pix] = sum_k A[oc, k] * B[k, pix],  k = (kh*3+kw)*64 + c, K=576=18*32
// 8 waves/block: wave t owns oc [16t,16t+16), 4 v8f accumulators cover 64
// pixels. 3 passes * 18 K-steps * 4 tiles = 216 bf16 wmma per wave.

typedef __attribute__((ext_vector_type(8)))  float  v8f;
typedef __attribute__((ext_vector_type(16))) __bf16 v16bf;
typedef __attribute__((ext_vector_type(8)))  __bf16 v8bf;

#define IN_FIX 4
#define CPF    16
#define OUTC   128
#define BATCH  32
#define HH     192
#define WW     192
#define CIN    64       // IN_FIX*CPF
#define HO     190
#define WO     190
#define NKS    18       // K-steps of 32: 9*64/32
#define NTILE  8        // OUTC/16
#define WTILE  64       // pixels per block
#define LDSROW 66       // WTILE + 2 halo columns
#define CPADB  72       // padded bf16 channel stride (144B: 16B-multiple, 36 dwords
                        // -> 36*col mod 64 is a bank permutation over the N-lanes)

// ---------------------------------------------------------------------------
// Repack weight[f, oc, j, kh, kw] into per-lane bf16 WMMA A fragments
// (hi plane then lo plane). ISA 16-bit A 16x32 layout:
//   lane<16 : elements 0..7 = K0..7,  8..15 = K16..23   (row M = lane)
//   lane>=16: elements 0..7 = K8..15, 8..15 = K24..31   (row M = lane-16)
// apre index: [sel(2)][t(8)][ks(18)][lane(32)] of v16bf (32B each).
// ---------------------------------------------------------------------------
__global__ void prep_weights_kernel(const float* __restrict__ w,
                                    v16bf* __restrict__ apre) {
    int idx = blockIdx.x * blockDim.x + threadIdx.x;
    if (idx >= 2 * NTILE * NKS * 32) return;
    int lane = idx & 31;
    int ks   = (idx >> 5) % NKS;
    int t    = (idx / (32 * NKS)) % NTILE;
    int sel  = idx / (32 * NKS * NTILE);     // 0 = hi plane, 1 = lo plane
    int oc   = t * 16 + (lane & 15);
    int khkw = ks >> 1, kb2 = ks & 1;        // k = khkw*64 + kb2*32 + koff
    int kh = khkw / 3, kw = khkw % 3;
    v16bf frag;
#pragma unroll
    for (int e = 0; e < 16; ++e) {
        int koff = ((e < 8) ? e : e + 8) + ((lane >> 4) << 3);
        int c = kb2 * 32 + koff;             // original input channel
        int f = c & 3, j = c >> 2;           // channel c = j*IN_FIX + f
        float wv = w[(((f * OUTC + oc) * CPF + j) * 9) + kh * 3 + kw];
        __bf16 hi = (__bf16)wv;
        frag[e] = sel ? (__bf16)(wv - (float)hi) : hi;
    }
    apre[idx] = frag;
}

// ---------------------------------------------------------------------------
// Main conv kernel. grid = (ceil(WO/64), HO, BATCH), block = 256 (8 waves).
// ---------------------------------------------------------------------------
__global__ __launch_bounds__(256)
void conv_wmma_bf16x3_kernel(const float* __restrict__ x,
                             const v16bf* __restrict__ apre,
                             const float* __restrict__ bias,
                             float* __restrict__ out) {
    __shared__ __bf16 xh[3 * LDSROW * CPADB];   // 28,512 B
    __shared__ __bf16 xl[3 * LDSROW * CPADB];   // 28,512 B

    const int wo0  = blockIdx.x * WTILE;
    const int ho   = blockIdx.y;
    const int b    = blockIdx.z;
    const int tid  = threadIdx.x;
    const int lane = tid & 31;
    const int wave = tid >> 5;

    // Fixation validity: marker is tensor[b, f, 0, 0] (NaN => invalid).
    float vmask[IN_FIX];
#pragma unroll
    for (int f = 0; f < IN_FIX; ++f) {
        float v = x[(size_t)(b * CIN + f) * (HH * WW)];
        vmask[f] = (v != v) ? 0.0f : 1.0f;
    }

    // Stage masked input tile, split into bf16 hi/lo planes: xs[kh][col][c].
    // col innermost over threads -> coalesced global row-segment reads.
    for (int e = tid; e < 3 * LDSROW * CIN; e += 256) {
        int col = e % LDSROW;
        int kh  = (e / LDSROW) % 3;
        int c   = e / (LDSROW * 3);
        int gcol = wo0 + col;
        float v = 0.0f;
        if (gcol < WW) {
            v = x[((b * CIN + c) * HH + (ho + kh)) * WW + gcol];
            v = (v != v) ? 0.0f : v;   // nan_to_num
            v *= vmask[c & 3];         // zero invalid fixation's channels
        }
        __bf16 hi = (__bf16)v;
        int si = (kh * LDSROW + col) * CPADB + c;
        xh[si] = hi;
        xl[si] = (__bf16)(v - (float)hi);
    }
    __syncthreads();

    const int t    = wave;                 // oc tile
    const int n    = lane & 15;            // N index within 16-pixel tile
    const int csub = (lane >> 4) << 3;     // B K: lane<16 -> K0..7,16..23, else K8..15,24..31

    v8f acc[4] = {};                       // 4 x (16 oc x 16 pixel) f32 tiles

    for (int pass = 0; pass < 3; ++pass) {
        // pass0: wh*xh   pass1: wh*xl   pass2: wl*xh
        const __bf16* bsrc = (pass == 1) ? xl : xh;
        const v16bf*  ap   = apre + ((pass == 2) ? (NTILE * NKS * 32) : 0)
                                  + (size_t)(t * NKS) * 32 + lane;
        for (int kh = 0; kh < 3; ++kh) {
            for (int kw = 0; kw < 3; ++kw) {
#pragma unroll
                for (int kb2 = 0; kb2 < 2; ++kb2) {   // two 32-channel blocks
                    v16bf a = *ap; ap += 32;          // ks = (kh*3+kw)*2 + kb2
                    int c0 = kb2 * 32 + csub;
#pragma unroll
                    for (int g = 0; g < 4; ++g) {
                        int si = (kh * LDSROW + (n + g * 16 + kw)) * CPADB + c0;
                        v8bf blo = *(const v8bf*)(bsrc + si);      // 16B aligned
                        v8bf bhi = *(const v8bf*)(bsrc + si + 16); // 16B aligned
                        v16bf bb = __builtin_shufflevector(
                            blo, bhi, 0, 1, 2, 3, 4, 5, 6, 7,
                                      8, 9, 10, 11, 12, 13, 14, 15);
                        acc[g] = __builtin_amdgcn_wmma_f32_16x16x32_bf16(
                            false, a, false, bb, (short)0, acc[g], false, false);
                    }
                }
            }
        }
    }

    // D layout: lane<16 -> M = r (acc element), lane>=16 -> M = r + 8.
    const int ocb = t * 16 + ((lane >> 4) << 3);
    float bsum[8];
#pragma unroll
    for (int r = 0; r < 8; ++r) {
        float s = 0.0f;
#pragma unroll
        for (int f = 0; f < IN_FIX; ++f) s += vmask[f] * bias[f * OUTC + ocb + r];
        bsum[r] = s;
    }
#pragma unroll
    for (int g = 0; g < 4; ++g) {
        int wo = wo0 + n + g * 16;
        if (wo < WO) {
#pragma unroll
            for (int r = 0; r < 8; ++r) {
                out[(((b * OUTC + ocb + r) * HO) + ho) * WO + wo] = acc[g][r] + bsum[r];
            }
        }
    }
}

// ---------------------------------------------------------------------------
extern "C" void kernel_launch(void* const* d_in, const int* in_sizes, int n_in,
                              void* d_out, int out_size, void* d_ws, size_t ws_size,
                              hipStream_t stream) {
    const float* tensor = (const float*)d_in[0]; // [32,64,192,192]
    const float* weight = (const float*)d_in[1]; // [4,128,16,3,3]
    const float* bias   = (const float*)d_in[2]; // [4,128]
    float* out  = (float*)d_out;                 // [32,128,190,190]
    v16bf* apre = (v16bf*)d_ws;                  // 294,912 B repacked bf16 hi/lo

    int prep_elems = 2 * NTILE * NKS * 32;       // 9216 fragments
    prep_weights_kernel<<<dim3((prep_elems + 255) / 256), dim3(256), 0, stream>>>(
        weight, apre);

    dim3 grid((WO + WTILE - 1) / WTILE, HO, BATCH);
    conv_wmma_bf16x3_kernel<<<grid, dim3(256), 0, stream>>>(tensor, apre, bias, out);
}
